// RNNModel_84258668413443
// MI455X (gfx1250) — matrix operand order repaired and checked by reading, not compiled, not run on hardware
//
#include <hip/hip_runtime.h>
#include <hip/hip_bf16.h>
#include <stdint.h>

// Problem constants (match reference).
#define TT  64
#define BB  32
#define VV  50000
#define HH  1024
#define LL  3
#define G4H (4 * HH)

typedef __attribute__((ext_vector_type(16))) __bf16 bf16x16;
typedef __attribute__((ext_vector_type(8)))  __bf16 bf16x8;
typedef __attribute__((ext_vector_type(8)))  float  f32x8;

__device__ __forceinline__ uint16_t f32_to_bf16_bits(float x) {
  union { float f; uint32_t u; } v; v.f = x;
  uint32_t r = v.u + 0x7FFFu + ((v.u >> 16) & 1u);  // round-to-nearest-even
  return (uint16_t)(r >> 16);
}

#define WMMA_BF16(A_, B_, C_)                                                 \
  __builtin_amdgcn_wmma_f32_16x16x32_bf16(false, (A_), false, (B_), (short)0, \
                                          (C_), false, false)

// A-fragment (16x32 bf16): 16 contiguous k per lane (k = 16h + s), loaded as two 8-elem chunks.
// Works for both global and LDS source pointers (addrspace inferred).
__device__ __forceinline__ bf16x16 load_afrag(const __bf16* __restrict__ p) {
  bf16x8 lo = *(const bf16x8*)(p);
  bf16x8 hi = *(const bf16x8*)(p + 8);
  bf16x16 a;
#pragma unroll
  for (int i = 0; i < 8; ++i) { a[i] = lo[i]; a[i + 8] = hi[i]; }
  return a;
}

// CDNA5 async global->LDS copy (ISA 10.x / CDNA5 ISA):
// per-lane 16B copy, LDS byte offset in a VGPR, tracked by ASYNCcnt.
__device__ __forceinline__ void async_copy_b128(uint32_t lds_off,
                                                const void* gaddr) {
  asm volatile("global_load_async_to_lds_b128 %0, %1, off"
               :: "v"(lds_off), "v"(gaddr) : "memory");
}
__device__ __forceinline__ void wait_async0() {
  asm volatile("s_wait_asynccnt 0x0" ::: "memory");
}

// ---------------------------------------------------------------------------
// LDS-tiled bf16 WMMA GEMM:  C[M,N] = A[M,K] * W[N,K]^T + bias[N]
// Block: 256 threads (8 waves), tile 128x128, K-panel 64, double-buffered LDS
// staged with async loads. Wave computes a 32x64 sub-tile (2x4 fragments).
// Requires: M % 128 == 0, K % 64 == 0, N % 16 == 0 (N tail < 128 handled).
// ---------------------------------------------------------------------------
#define BM 128
#define BN 128
#define BK 64
#define LDSTRIDE 72  // 64 + 8-elem pad: rows 0..15 land on disjoint bank groups

__global__ __launch_bounds__(256)
void gemm_lds_bf16(const uint16_t* __restrict__ Abits,
                   const uint16_t* __restrict__ Wbits,
                   const float* __restrict__ bias,
                   float* __restrict__ C,
                   int M, int N, int K) {
  __shared__ __align__(16) uint16_t sA[2][BM * LDSTRIDE];
  __shared__ __align__(16) uint16_t sB[2][BN * LDSTRIDE];

  const __bf16* A = reinterpret_cast<const __bf16*>(Abits);
  const __bf16* W = reinterpret_cast<const __bf16*>(Wbits);

  const int tid  = threadIdx.x;
  const int lane = tid & 31;
  const int w    = tid >> 5;           // wave 0..7
  const int Nb   = (N + BN - 1) / BN;
  const int m0blk = (int)(blockIdx.x / Nb) * BM;
  const int n0blk = (int)(blockIdx.x % Nb) * BN;

  // Wave sub-tile origin within the block tile: 4x2 arrangement of 32x64.
  const int wm = (w & 3) * 32;
  const int wn = (w >> 2) * 64;

  // Fragment lane addressing (contiguous k: lane half h holds k 16h..16h+15).
  const int fm  = lane & 15;
  const int kAo = (lane >> 4) << 4;    // 0 or 16
  const int kBo = (lane >> 4) << 4;    // 0 or 16

  // Stage one K-panel (A: 128x64, B: 128x64) into LDS buffer `buf`.
  // 8 chunks of 16B per row; 256 threads x 4 chunks per matrix.
  auto stage_panel = [&](int kp, int buf) {
#pragma unroll
    for (int j = 0; j < 4; ++j) {
      const int ch = tid + 256 * j;
      const int r  = ch >> 3;            // 0..127
      const int ce = (ch & 7) * 8;       // elem col 0..56
      const int ldsoff = r * LDSTRIDE + ce;
      async_copy_b128((uint32_t)(uintptr_t)&sA[buf][ldsoff],
                      A + (size_t)(m0blk + r) * K + kp * BK + ce);
      int nr = n0blk + r;
      if (nr > N - 1) nr = N - 1;        // clamp; lands only in skipped frags
      async_copy_b128((uint32_t)(uintptr_t)&sB[buf][ldsoff],
                      W + (size_t)nr * K + kp * BK + ce);
    }
  };

  f32x8 acc[2][4] = {};

  stage_panel(0, 0);
  wait_async0();
  __syncthreads();

  const int np = K / BK;
  for (int kp = 0; kp < np; ++kp) {
    const int cur = kp & 1;
    if (kp + 1 < np) stage_panel(kp + 1, cur ^ 1);  // overlap DMA with compute

#pragma unroll
    for (int ks = 0; ks < BK; ks += 32) {
      // Materialize all six fragments first so the 12 ds_load_b128 issue as
      // one clause and the dscnt waits stagger across the 8-WMMA burst.
      bf16x16 a0 =
          load_afrag((const __bf16*)&sA[cur][(wm + fm) * LDSTRIDE + ks + kAo]);
      bf16x16 a1 = load_afrag(
          (const __bf16*)&sA[cur][(wm + 16 + fm) * LDSTRIDE + ks + kAo]);
      bf16x16 b0 =
          *(const bf16x16*)&sB[cur][(wn + 0 + fm) * LDSTRIDE + ks + kBo];
      bf16x16 b1 =
          *(const bf16x16*)&sB[cur][(wn + 16 + fm) * LDSTRIDE + ks + kBo];
      bf16x16 b2 =
          *(const bf16x16*)&sB[cur][(wn + 32 + fm) * LDSTRIDE + ks + kBo];
      bf16x16 b3 =
          *(const bf16x16*)&sB[cur][(wn + 48 + fm) * LDSTRIDE + ks + kBo];

      acc[0][0] = WMMA_BF16(a0, b0, acc[0][0]);
      acc[1][0] = WMMA_BF16(a1, b0, acc[1][0]);
      acc[0][1] = WMMA_BF16(a0, b1, acc[0][1]);
      acc[1][1] = WMMA_BF16(a1, b1, acc[1][1]);
      acc[0][2] = WMMA_BF16(a0, b2, acc[0][2]);
      acc[1][2] = WMMA_BF16(a1, b2, acc[1][2]);
      acc[0][3] = WMMA_BF16(a0, b3, acc[0][3]);
      acc[1][3] = WMMA_BF16(a1, b3, acc[1][3]);
    }

    wait_async0();     // own async loads of next panel complete
    __syncthreads();   // everyone done loading next / reading current
  }

  // Epilogue: lane holds col n0+fm; rows m0 + 8*(lane/16) + i across 8 regs.
  const int mhalf = (lane >> 4) << 3;
#pragma unroll
  for (int mf = 0; mf < 2; ++mf) {
#pragma unroll
    for (int f = 0; f < 4; ++f) {
      const int n0 = n0blk + wn + f * 16;
      if (n0 >= N) continue;  // uniform; decoder tail only
      const int n = n0 + fm;
      const float bv = bias ? bias[n] : 0.0f;
      const int mbase = m0blk + wm + mf * 16 + mhalf;
#pragma unroll
      for (int i = 0; i < 8; ++i)
        C[(size_t)(mbase + i) * N + n] = acc[mf][f][i] + bv;
    }
  }
}

// ---------------------------------------------------------------------------
// Direct bf16 WMMA GEMM (for the small recurrent step, M=32, L2-resident W):
// one wave -> 16x64 tile, register double-buffered. N % 64 == 0, K % 32 == 0.
// ---------------------------------------------------------------------------
__global__ void gemm_direct_bf16(const uint16_t* __restrict__ Abits,
                                 const uint16_t* __restrict__ Wbits,
                                 const float* __restrict__ addend,
                                 float* __restrict__ C,
                                 int M, int N, int K) {
  const __bf16* A = reinterpret_cast<const __bf16*>(Abits);
  const __bf16* W = reinterpret_cast<const __bf16*>(Wbits);

  const int lane = threadIdx.x & 31;
  const int wave = (int)((blockIdx.x * blockDim.x + threadIdx.x) >> 5);
  const int Mt = M >> 4;
  const int Nt = N >> 6;
  if (wave >= Mt * Nt) return;  // wave-uniform exit
  const int m0 = (wave % Mt) << 4;
  const int n0base = (wave / Mt) << 6;

  const int mrow = m0 + (lane & 15);
  const int kA   = (lane >> 4) << 4;
  const int kB   = (lane >> 4) << 4;
  const int ncol = lane & 15;

  const __bf16* Ap = A + (size_t)mrow * K + kA;
  const __bf16* Bp = W + (size_t)(n0base + ncol) * K + kB;
  const size_t bstep = (size_t)16 * K;

  f32x8 acc0 = {}, acc1 = {}, acc2 = {}, acc3 = {};

  bf16x16 ac = load_afrag(Ap);
  bf16x16 b0 = *(const bf16x16*)(Bp);
  bf16x16 b1 = *(const bf16x16*)(Bp + bstep);
  bf16x16 b2 = *(const bf16x16*)(Bp + 2 * bstep);
  bf16x16 b3 = *(const bf16x16*)(Bp + 3 * bstep);
  for (int k0 = 32; k0 < K; k0 += 32) {
    bf16x16 an  = load_afrag(Ap + k0);
    bf16x16 bn0 = *(const bf16x16*)(Bp + k0);
    bf16x16 bn1 = *(const bf16x16*)(Bp + bstep + k0);
    bf16x16 bn2 = *(const bf16x16*)(Bp + 2 * bstep + k0);
    bf16x16 bn3 = *(const bf16x16*)(Bp + 3 * bstep + k0);
    acc0 = WMMA_BF16(ac, b0, acc0);
    acc1 = WMMA_BF16(ac, b1, acc1);
    acc2 = WMMA_BF16(ac, b2, acc2);
    acc3 = WMMA_BF16(ac, b3, acc3);
    ac = an; b0 = bn0; b1 = bn1; b2 = bn2; b3 = bn3;
  }
  acc0 = WMMA_BF16(ac, b0, acc0);
  acc1 = WMMA_BF16(ac, b1, acc1);
  acc2 = WMMA_BF16(ac, b2, acc2);
  acc3 = WMMA_BF16(ac, b3, acc3);

  const int mbase = m0 + ((lane >> 4) << 3);
#pragma unroll
  for (int f = 0; f < 4; ++f) {
    const int n = n0base + (f << 4) + ncol;
    const f32x8 a = (f == 0) ? acc0 : (f == 1) ? acc1 : (f == 2) ? acc2 : acc3;
#pragma unroll
    for (int i = 0; i < 8; ++i) {
      const size_t row = (size_t)(mbase + i) * N + n;
      float v = a[i];
      if (addend) v += addend[row];
      C[row] = v;
    }
  }
}

// ---------------------------------------------------------------------------
// Small helper kernels
// ---------------------------------------------------------------------------
__global__ void cvt_f32_bf16_kernel(const float* __restrict__ src,
                                    uint16_t* __restrict__ dst, long long n) {
  long long i = (long long)blockIdx.x * blockDim.x + threadIdx.x;
  if (i < n) dst[i] = f32_to_bf16_bits(src[i]);
}

__global__ void embed_kernel(const int* __restrict__ tokens,
                             const float* __restrict__ embW,
                             uint16_t* __restrict__ xout) {
  long long i = (long long)blockIdx.x * blockDim.x + threadIdx.x;
  if (i >= (long long)TT * BB * HH) return;
  int col = (int)(i % HH);
  int row = (int)(i / HH);
  int tok = tokens[row];
  xout[i] = f32_to_bf16_bits(embW[(long long)tok * HH + col]);
}

__global__ void bias_comb_kernel(const float* __restrict__ a,
                                 const float* __restrict__ b,
                                 float* __restrict__ out) {
  int i = blockIdx.x * blockDim.x + threadIdx.x;
  if (i < G4H) out[i] = a[i] + b[i];
}

__global__ void init_state_kernel(const float* __restrict__ h0,
                                  const float* __restrict__ c0,
                                  uint16_t* __restrict__ hb,
                                  float* __restrict__ c) {
  int i = blockIdx.x * blockDim.x + threadIdx.x;
  if (i < BB * HH) { hb[i] = f32_to_bf16_bits(h0[i]); c[i] = c0[i]; }
}

// gates order (torch LSTMCell): i, f, g, o blocks of H.
__global__ void lstm_pointwise_kernel(const float* __restrict__ gates,
                                      float* __restrict__ c,
                                      uint16_t* __restrict__ hb,
                                      uint16_t* __restrict__ xout, int t) {
  int i = blockIdx.x * blockDim.x + threadIdx.x;
  if (i >= BB * HH) return;
  int b = i / HH, h = i - b * HH;
  const float* g = gates + (long long)b * G4H;
  float gi = 1.0f / (1.0f + __expf(-g[h]));
  float gf = 1.0f / (1.0f + __expf(-g[HH + h]));
  float gg = tanhf(g[2 * HH + h]);
  float go = 1.0f / (1.0f + __expf(-g[3 * HH + h]));
  float cn = gf * c[i] + gi * gg;
  c[i] = cn;
  uint16_t hbits = f32_to_bf16_bits(go * tanhf(cn));
  hb[i] = hbits;
  xout[((long long)t * BB + b) * HH + h] = hbits;
}

// ---------------------------------------------------------------------------
// Host side
// ---------------------------------------------------------------------------
static inline void launch_gemm_lds(const void* A, const void* W,
                                   const float* bias, float* C, int M, int N,
                                   int K, hipStream_t s) {
  unsigned blocks = (unsigned)((M / BM) * ((N + BN - 1) / BN));
  hipLaunchKernelGGL(gemm_lds_bf16, dim3(blocks), dim3(256), 0, s,
                     (const uint16_t*)A, (const uint16_t*)W, bias, C, M, N, K);
}

static inline void launch_gemm_direct(const void* A, const void* W,
                                      const float* addend, float* C, int M,
                                      int N, int K, hipStream_t s) {
  long long waves = (long long)(M / 16) * (N / 64);
  unsigned grid = (unsigned)((waves * 32 + 255) / 256);
  hipLaunchKernelGGL(gemm_direct_bf16, dim3(grid), dim3(256), 0, s,
                     (const uint16_t*)A, (const uint16_t*)W, addend, C, M, N, K);
}

extern "C" void kernel_launch(void* const* d_in, const int* in_sizes, int n_in,
                              void* d_out, int out_size, void* d_ws, size_t ws_size,
                              hipStream_t stream) {
  (void)in_sizes; (void)n_in; (void)out_size; (void)ws_size;
  const int*   tokens = (const int*)  d_in[0];  // [T,B]
  const float* h0     = (const float*)d_in[1];  // [L,B,H]
  const float* c0     = (const float*)d_in[2];  // [L,B,H]
  const float* embW   = (const float*)d_in[3];  // [V,H]
  const float* Wih    = (const float*)d_in[4];  // [L,4H,H]
  const float* Whh    = (const float*)d_in[5];  // [L,4H,H]
  const float* bih    = (const float*)d_in[6];  // [L,4H]
  const float* bhh    = (const float*)d_in[7];  // [L,4H]
  const float* decW   = (const float*)d_in[8];  // [V,H]
  const float* decb   = (const float*)d_in[9];  // [V]
  float* logits = (float*)d_out;                // [T,B,V]

  // Workspace layout (bytes). Total ~154.3 MiB; all chunks 32B+ aligned.
  char* ws = (char*)d_ws;
  const size_t OFF_XBUF0 = 0;
  const size_t OFF_XBUF1 = OFF_XBUF0 + (size_t)TT * BB * HH * 2;   //  4 MiB
  const size_t OFF_XG    = OFF_XBUF1 + (size_t)TT * BB * HH * 2;   //  4 MiB
  const size_t OFF_WIH   = OFF_XG    + (size_t)TT * BB * G4H * 4;  // 32 MiB
  const size_t OFF_WHH   = OFF_WIH   + (size_t)G4H * HH * 2;       //  8 MiB
  const size_t OFF_DECW  = OFF_WHH   + (size_t)G4H * HH * 2;       //  8 MiB
  const size_t OFF_HBF   = OFF_DECW  + (size_t)VV * HH * 2;        // 102.4 MB
  const size_t OFF_C     = OFF_HBF   + (size_t)BB * HH * 2;
  const size_t OFF_GATES = OFF_C     + (size_t)BB * HH * 4;
  const size_t OFF_BIAS  = OFF_GATES + (size_t)BB * G4H * 4;

  uint16_t* xbuf0   = (uint16_t*)(ws + OFF_XBUF0);
  uint16_t* xbuf1   = (uint16_t*)(ws + OFF_XBUF1);
  float*    xg      = (float*)   (ws + OFF_XG);
  uint16_t* wih_bf  = (uint16_t*)(ws + OFF_WIH);
  uint16_t* whh_bf  = (uint16_t*)(ws + OFF_WHH);
  uint16_t* decw_bf = (uint16_t*)(ws + OFF_DECW);
  uint16_t* h_bf    = (uint16_t*)(ws + OFF_HBF);
  float*    c_f32   = (float*)   (ws + OFF_C);
  float*    gates   = (float*)   (ws + OFF_GATES);
  float*    bias_c  = (float*)   (ws + OFF_BIAS);

  // 1) Embedding gather + bf16 convert.
  {
    long long n = (long long)TT * BB * HH;
    hipLaunchKernelGGL(embed_kernel, dim3((unsigned)((n + 255) / 256)),
                       dim3(256), 0, stream, tokens, embW, xbuf0);
  }

  // 2) dec_W -> bf16 once (102.4 MB; L2-resident for the decoder GEMM).
  {
    long long n = (long long)VV * HH;
    hipLaunchKernelGGL(cvt_f32_bf16_kernel, dim3((unsigned)((n + 255) / 256)),
                       dim3(256), 0, stream, decW, decw_bf, n);
  }

  uint16_t* xin = xbuf0;
  uint16_t* xout = xbuf1;

  for (int l = 0; l < LL; ++l) {
    const long long wsz = (long long)G4H * HH;
    hipLaunchKernelGGL(cvt_f32_bf16_kernel, dim3((unsigned)((wsz + 255) / 256)),
                       dim3(256), 0, stream, Wih + (long long)l * wsz, wih_bf, wsz);
    hipLaunchKernelGGL(cvt_f32_bf16_kernel, dim3((unsigned)((wsz + 255) / 256)),
                       dim3(256), 0, stream, Whh + (long long)l * wsz, whh_bf, wsz);
    hipLaunchKernelGGL(bias_comb_kernel, dim3((G4H + 255) / 256), dim3(256), 0,
                       stream, bih + l * G4H, bhh + l * G4H, bias_c);

    // Input projection: xg[T*B,4H] = x * Wih^T + (b_ih + b_hh)   (LDS-tiled)
    launch_gemm_lds(xin, wih_bf, bias_c, xg, TT * BB, G4H, HH, stream);

    hipLaunchKernelGGL(init_state_kernel, dim3((BB * HH + 255) / 256), dim3(256),
                       0, stream, h0 + (long long)l * BB * HH,
                       c0 + (long long)l * BB * HH, h_bf, c_f32);

    for (int t = 0; t < TT; ++t) {
      // gates[B,4H] = h * Whh^T + xg[t]   (small M, L2-resident W: direct)
      launch_gemm_direct(h_bf, whh_bf, xg + (long long)t * BB * G4H, gates,
                         BB, G4H, HH, stream);
      hipLaunchKernelGGL(lstm_pointwise_kernel, dim3((BB * HH + 255) / 256),
                         dim3(256), 0, stream, gates, c_f32, h_bf, xout, t);
    }
    uint16_t* tmp = xin; xin = xout; xout = tmp;
  }

  // 3) Decoder: logits[T*B, V] = x * dec_W^T + dec_b  (dominant GEMM, LDS-tiled)
  launch_gemm_lds(xin, decw_bf, decb, logits, TT * BB, VV, HH, stream);
}
